// SwinTransformerBlock_79688823210191
// MI455X (gfx1250) — matrix operand order repaired and checked by reading, not compiled, not run on hardware
//
#include <hip/hip_runtime.h>
#include <hip/hip_bf16.h>
#include <math.h>

// ---- problem constants ----
#define LL      3136            // H*W
#define BB      32
#define CC      256
#define NHEADS  8
#define HD      32
#define NWIN    64              // windows per batch image
#define NTOK    49
#define SHIFT_  24
#define HIDDEN  1024
#define MROWS   (BB*LL)         // 100352 tokens
#define QKSCALE 0.17677669529663687f   // (256/8)^-0.5

typedef __attribute__((ext_vector_type(16))) __bf16 v16bf;
typedef __attribute__((ext_vector_type(8)))  float  v8f;

static __device__ __forceinline__ v8f wmma_bf16(v16bf a, v16bf b, v8f c) {
  // D = A(16x32) * B(32x16) + C, f32 accum
  return __builtin_amdgcn_wmma_f32_16x16x32_bf16(false, a, false, b, (short)0, c, false, false);
}

static __device__ __forceinline__ v8f zero8() {
  v8f z;
#pragma unroll
  for (int i = 0; i < 8; i++) z[i] = 0.0f;
  return z;
}

// Load the 16-bit A-matrix fragment (16x32, MxK) in contiguous K order (not the ISA table order):
// lane: m = lane&15, half = lane>>4; elem i -> K = half*16+i
static __device__ __forceinline__ v16bf load_a_frag(const __bf16* rowK0, int half) {
  v16bf a;
#pragma unroll
  for (int i = 0; i < 16; i++) {
    a[i] = rowK0[half * 16 + i];
  }
  return a;
}

// CDNA5 async copy: 16B per lane, global -> LDS, tracked by ASYNCcnt.
static __device__ __forceinline__ void async_g2l_b128(unsigned lds_addr, const void* gptr) {
  unsigned long long ga = (unsigned long long)(size_t)gptr;
  asm volatile("global_load_async_to_lds_b128 %0, %1, off"
               :: "v"(lds_addr), "v"(ga) : "memory");
}
static __device__ __forceinline__ void async_wait0() {
  asm volatile("s_wait_asynccnt 0x0" ::: "memory");
}
static __device__ __forceinline__ unsigned lds_off(const void* p) {
  return (unsigned)(size_t)p;   // low 32 bits of generic LDS pointer = LDS offset
}

// ---------------- weight packer: f32 (K x N, row major) -> B-fragment-ordered bf16 ----------------
// Fragment (kt,nt): 512 bf16, ordered [lane][i]; lane: n = lane&15, half = lane>>4; K = kt*32 + half*16 + i
__global__ __launch_bounds__(256) void pack_w_kernel(const float* __restrict__ W,
                                                     __bf16* __restrict__ out, int K, int N) {
  long idx = (long)blockIdx.x * 256 + threadIdx.x;
  long total = (long)K * N;
  if (idx >= total) return;
  int ntiles = N >> 4;
  int frag = (int)(idx >> 9);
  int e    = (int)(idx & 511);
  int lane = e >> 4, i = e & 15;
  int kt = frag / ntiles, nt = frag % ntiles;
  int n = lane & 15, half = lane >> 4;
  int k = kt * 32 + half * 16 + i;
  out[idx] = (__bf16)W[(long)k * N + nt * 16 + n];
}

// ---------------- layernorm (+ optional forward roll along L), f32 in -> bf16 out ----------------
__global__ __launch_bounds__(256) void ln_kernel(const float* __restrict__ x,
                                                 const float* __restrict__ w,
                                                 const float* __restrict__ b,
                                                 __bf16* __restrict__ out, int roll) {
  int row  = blockIdx.x * 8 + (threadIdx.x >> 5);
  int lane = threadIdx.x & 31;
  int bi = row / LL, t = row % LL;
  int src = roll ? (bi * LL + ((t - roll) % LL + LL) % LL) : row;
  const float* xr = x + (size_t)src * CC;
  float v[8], s = 0.f, s2 = 0.f;
#pragma unroll
  for (int i = 0; i < 8; i++) { float vv = xr[i * 32 + lane]; v[i] = vv; s += vv; s2 += vv * vv; }
#pragma unroll
  for (int m = 1; m < 32; m <<= 1) { s += __shfl_xor(s, m, 32); s2 += __shfl_xor(s2, m, 32); }
  float mean = s * (1.0f / CC);
  float var  = s2 * (1.0f / CC) - mean * mean;
  float rstd = rsqrtf(var + 1e-5f);
#pragma unroll
  for (int i = 0; i < 8; i++) {
    int c = i * 32 + lane;
    out[(size_t)row * CC + c] = (__bf16)((v[i] - mean) * rstd * w[c] + b[c]);
  }
}

// ---------------- QKV GEMM: block stages 32x256 A in LDS (async), 8 waves x (2 mt x 48 nt) ----------------
__global__ __launch_bounds__(256) void qkv_gemm_kernel(const __bf16* __restrict__ A,
                                                       const __bf16* __restrict__ Wp,
                                                       const float* __restrict__ bias,
                                                       __bf16* __restrict__ out) {
  __shared__ __align__(16) __bf16 As[32][256];   // 16 KB
  int tid = threadIdx.x;
  int wave = tid >> 5, lane = tid & 31;
  int half = lane >> 4, m = lane & 15;
  int rowbase = blockIdx.x * 32;

  {  // async stage: 1024 x 16B chunks, 4 per thread (flat copy, layouts match)
    unsigned lbase = lds_off(&As[0][0]);
    const char* gbase = (const char*)(A + (size_t)rowbase * CC);
#pragma unroll
    for (int c4 = 0; c4 < 4; c4++) {
      int c = tid * 4 + c4;
      async_g2l_b128(lbase + c * 16, gbase + (size_t)c * 16);
    }
    async_wait0();
  }
  __syncthreads();

  int mt = wave & 1;
  int nbase = (wave >> 1) * 12;                  // 12 N-tiles per wave, 3 groups of 4
  const __bf16* arow = &As[mt * 16 + m][0];
  for (int g = 0; g < 3; g++) {
    v8f acc[4];
#pragma unroll
    for (int j = 0; j < 4; j++) acc[j] = zero8();
    for (int kt = 0; kt < 8; kt++) {
      v16bf af = load_a_frag(arow + kt * 32, half);
      const __bf16* wb = Wp + ((size_t)(kt * 48 + nbase + g * 4) * 512) + lane * 16;
      __builtin_prefetch((const void*)(wb + (size_t)48 * 512), 0, 1);   // next K-panel
#pragma unroll
      for (int j = 0; j < 4; j++)
        acc[j] = wmma_bf16(af, *(const v16bf*)(wb + (size_t)j * 512), acc[j]);
    }
    size_t rbase = (size_t)(rowbase + mt * 16 + half * 8);
#pragma unroll
    for (int j = 0; j < 4; j++) {
      int col = (nbase + g * 4 + j) * 16 + m;
      float bb = bias[col];
#pragma unroll
      for (int r = 0; r < 8; r++)
        out[(rbase + r) * 768 + col] = (__bf16)(acc[j][r] + bb);
    }
  }
}

// ---------------- Attention: one block (4 waves) per (window, head); V staged via async LDS ----------------
__global__ __launch_bounds__(128) void attn_kernel(const __bf16* __restrict__ qkv,
                                                   const float* __restrict__ bias_tb,
                                                   const int* __restrict__ rel,
                                                   __bf16* __restrict__ attn_out) {
  __shared__ __align__(16) __bf16 Ps[4][16][64];   // 8 KB probability tiles
  __shared__ __align__(16) __bf16 Vs[64][32];      // 4 KB V tile (rows >=49 zero)
  int wh   = blockIdx.x;
  int win  = wh >> 3, head = wh & 7;
  int winb = win & (NWIN - 1);
  int wave = threadIdx.x >> 5, lane = threadIdx.x & 31;
  int half = lane >> 4, m = lane & 15;
  int tid  = threadIdx.x;

  const __bf16* qbase = qkv + (size_t)win * NTOK * 768 + head * HD;
  const __bf16* kbase = qbase + 256;
  const __bf16* vbase = qbase + 512;

  // zero pad rows (disjoint from async-written rows)
  for (int e = tid; e < 15 * 32; e += 128) Vs[NTOK + e / 32][e % 32] = (__bf16)0.0f;
  // async stage V: 49 rows x 64B = 196 x 16B chunks
  {
    unsigned lbase = lds_off(&Vs[0][0]);
    int c = tid;                     // 0..127
    async_g2l_b128(lbase + c * 16, (const char*)vbase + (size_t)(c >> 2) * 1536 + (c & 3) * 16);
    int c2 = tid + 128;
    if (c2 < 196)
      async_g2l_b128(lbase + c2 * 16, (const char*)vbase + (size_t)(c2 >> 2) * 1536 + (c2 & 3) * 16);
    async_wait0();
  }

  // ---- S = q @ k^T (rows wave*16 .. +15; K = 32, one WMMA step) ----
  v16bf af;
  {
    int qrow = wave * 16 + m;
    if (qrow < NTOK) {
      af = load_a_frag(qbase + (size_t)qrow * 768, half);
    } else {
#pragma unroll
      for (int i = 0; i < 16; i++) af[i] = (__bf16)0.0f;
    }
  }
  v8f acc[4];
#pragma unroll
  for (int nt = 0; nt < 4; nt++) {
    v16bf bf;
    int tn = nt * 16 + m;                    // key token = B column
    if (tn < NTOK) {
      const __bf16* rp = kbase + (size_t)tn * 768 + half * 16;
#pragma unroll
      for (int i = 0; i < 16; i++) bf[i] = rp[i];
    } else {
#pragma unroll
      for (int i = 0; i < 16; i++) bf[i] = (__bf16)0.0f;
    }
    acc[nt] = wmma_bf16(af, bf, zero8());
  }

  // ---- bias + softmax (row in one 16-lane half; cols across 4 accumulators) ----
#pragma unroll
  for (int r = 0; r < 8; r++) {
    int qrow = wave * 16 + half * 8 + r;
    int qc = qrow < NTOK ? qrow : NTOK - 1;
    float sv[4];
    float mx = -1e30f;
#pragma unroll
    for (int nt = 0; nt < 4; nt++) {
      int kcol = nt * 16 + m;
      float val;
      if (kcol < NTOK) {
        int bi = rel[((size_t)winb * NTOK + qc) * NTOK + kcol];
        val = acc[nt][r] * QKSCALE + bias_tb[(size_t)bi * NHEADS + head];
      } else {
        val = -1e30f;
      }
      sv[nt] = val;
      mx = fmaxf(mx, val);
    }
#pragma unroll
    for (int msk = 1; msk < 16; msk <<= 1) mx = fmaxf(mx, __shfl_xor(mx, msk, 32));
    float se = 0.f;
#pragma unroll
    for (int nt = 0; nt < 4; nt++) { sv[nt] = __expf(sv[nt] - mx); se += sv[nt]; }
#pragma unroll
    for (int msk = 1; msk < 16; msk <<= 1) se += __shfl_xor(se, msk, 32);
    float inv = 1.0f / se;
#pragma unroll
    for (int nt = 0; nt < 4; nt++)
      Ps[wave][half * 8 + r][nt * 16 + m] = (__bf16)(sv[nt] * inv);
  }
  __syncthreads();   // Ps + Vs visible to all waves

  // ---- out = P @ v   (K = 49 padded to 64 -> 2 WMMA K-steps; N = 32 -> 2 tiles) ----
  v8f oacc[2];
#pragma unroll
  for (int nt = 0; nt < 2; nt++) oacc[nt] = zero8();
  for (int kt = 0; kt < 2; kt++) {
    v16bf pa = load_a_frag(&Ps[wave][m][kt * 32], half);
#pragma unroll
    for (int nt = 0; nt < 2; nt++) {
      v16bf vb;
#pragma unroll
      for (int i = 0; i < 16; i++)
        vb[i] = Vs[kt * 32 + half * 16 + i][nt * 16 + m];
      oacc[nt] = wmma_bf16(pa, vb, oacc[nt]);
    }
  }
#pragma unroll
  for (int nt = 0; nt < 2; nt++) {
#pragma unroll
    for (int r = 0; r < 8; r++) {
      int orow = wave * 16 + half * 8 + r;
      if (orow < NTOK)
        attn_out[((size_t)win * NTOK + orow) * CC + head * HD + nt * 16 + m] = (__bf16)oacc[nt][r];
    }
  }
}

// ---------------- proj GEMM + bias + reverse roll + residual -> f32 x1 (in d_out) ----------------
__global__ __launch_bounds__(256) void proj_gemm_kernel(const __bf16* __restrict__ A,
                                                        const __bf16* __restrict__ Wp,
                                                        const float* __restrict__ bias,
                                                        const float* __restrict__ x,
                                                        float* __restrict__ out) {
  __shared__ __align__(16) __bf16 As[32][256];   // 16 KB
  int tid = threadIdx.x;
  int wave = tid >> 5, lane = tid & 31;
  int half = lane >> 4, m = lane & 15;
  int rowbase = blockIdx.x * 32;

  {
    unsigned lbase = lds_off(&As[0][0]);
    const char* gbase = (const char*)(A + (size_t)rowbase * CC);
#pragma unroll
    for (int c4 = 0; c4 < 4; c4++) {
      int c = tid * 4 + c4;
      async_g2l_b128(lbase + c * 16, gbase + (size_t)c * 16);
    }
    async_wait0();
  }
  __syncthreads();

  int mt = wave & 1;
  int ng = wave >> 1;                            // 4 N-tiles per wave
  const __bf16* arow = &As[mt * 16 + m][0];
  v8f acc[4];
#pragma unroll
  for (int j = 0; j < 4; j++) acc[j] = zero8();
  for (int kt = 0; kt < 8; kt++) {
    v16bf af = load_a_frag(arow + kt * 32, half);
    const __bf16* wb = Wp + ((size_t)(kt * 16 + ng * 4) * 512) + lane * 16;
    __builtin_prefetch((const void*)(wb + (size_t)16 * 512), 0, 1);
#pragma unroll
    for (int j = 0; j < 4; j++)
      acc[j] = wmma_bf16(af, *(const v16bf*)(wb + (size_t)j * 512), acc[j]);
  }
#pragma unroll
  for (int r = 0; r < 8; r++) {
    int rrow = rowbase + mt * 16 + half * 8 + r;   // row in rolled layout
    int bi = rrow / LL, t = rrow % LL;
    int dst = bi * LL + ((t - SHIFT_) + LL) % LL;  // reverse roll
#pragma unroll
    for (int j = 0; j < 4; j++) {
      int col = (ng * 4 + j) * 16 + m;
      size_t o = (size_t)dst * CC + col;
      out[o] = acc[j][r] + bias[col] + x[o];
    }
  }
}

// ---------------- fused MLP: async-staged A, hidden tile lives entirely in LDS ----------------
__global__ __launch_bounds__(256) void mlp_kernel(const __bf16* __restrict__ h2,
                                                  const __bf16* __restrict__ W1p,
                                                  const float* __restrict__ b1,
                                                  const __bf16* __restrict__ W2p,
                                                  const float* __restrict__ b2,
                                                  float* __restrict__ out) {
  __shared__ __align__(16) __bf16 As[16][256];    // 8 KB input rows
  __shared__ __align__(16) __bf16 Hs[16][HIDDEN]; // 32 KB hidden (post-GELU)
  int tid = threadIdx.x;
  int wave = tid >> 5, lane = tid & 31;
  int half = lane >> 4, m = lane & 15;
  int rowbase = blockIdx.x * 16;

  {  // async stage A: 512 x 16B chunks, 2 per thread
    unsigned lbase = lds_off(&As[0][0]);
    const char* gbase = (const char*)(h2 + (size_t)rowbase * CC);
#pragma unroll
    for (int c2 = 0; c2 < 2; c2++) {
      int c = tid * 2 + c2;
      async_g2l_b128(lbase + c * 16, gbase + (size_t)c * 16);
    }
    async_wait0();
  }
  __syncthreads();

  const __bf16* arow = &As[m][0];

  // phase 1: H = gelu(As @ W1 + b1) ; 64 N-tiles over 8 waves
#pragma unroll
  for (int j = 0; j < 8; j++) {
    int nt = wave * 8 + j;
    v8f acc = zero8();
    for (int kt = 0; kt < 8; kt++) {
      v16bf af = load_a_frag(arow + kt * 32, half);
      const v16bf* bp = (const v16bf*)(W1p + ((size_t)(kt * 64 + nt) * 512) + lane * 16);
      acc = wmma_bf16(af, *bp, acc);
    }
    int col = nt * 16 + m;
    float bb = b1[col];
#pragma unroll
    for (int r = 0; r < 8; r++) {
      float vv = acc[r] + bb;
      float g = 0.5f * vv * (1.0f + erff(vv * 0.70710678118654752f));   // exact GELU
      Hs[half * 8 + r][col] = (__bf16)g;
    }
  }
  __syncthreads();

  // phase 2: out += Hs @ W2 + b2 ; K = 1024 (32 steps), 16 N-tiles over 8 waves
#pragma unroll
  for (int j = 0; j < 2; j++) {
    int nt = wave * 2 + j;
    v8f acc = zero8();
    for (int kt = 0; kt < 32; kt++) {
      v16bf af = load_a_frag(&Hs[m][kt * 32], half);
      const __bf16* wb = W2p + ((size_t)(kt * 16 + nt) * 512) + lane * 16;
      __builtin_prefetch((const void*)(wb + (size_t)16 * 512), 0, 1);
      acc = wmma_bf16(af, *(const v16bf*)wb, acc);
    }
    int col = nt * 16 + m;
    float bb = b2[col];
#pragma unroll
    for (int r = 0; r < 8; r++) {
      size_t o = (size_t)(rowbase + half * 8 + r) * CC + col;
      out[o] = out[o] + acc[r] + bb;          // residual: out currently holds x1
    }
  }
}

extern "C" void kernel_launch(void* const* d_in, const int* in_sizes, int n_in,
                              void* d_out, int out_size, void* d_ws, size_t ws_size,
                              hipStream_t stream) {
  (void)in_sizes; (void)n_in; (void)out_size; (void)ws_size;
  const float* x       = (const float*)d_in[0];
  const float* qkv_w   = (const float*)d_in[1];
  const float* qkv_b   = (const float*)d_in[2];
  const float* proj_w  = (const float*)d_in[3];
  const float* proj_b  = (const float*)d_in[4];
  const float* bias_tb = (const float*)d_in[5];
  const float* n1w     = (const float*)d_in[6];
  const float* n1b     = (const float*)d_in[7];
  const float* n2w     = (const float*)d_in[8];
  const float* n2b     = (const float*)d_in[9];
  const float* fc1_w   = (const float*)d_in[10];
  const float* fc1_b   = (const float*)d_in[11];
  const float* fc2_w   = (const float*)d_in[12];
  const float* fc2_b   = (const float*)d_in[13];
  const int*   rel     = (const int*)d_in[14];
  float* out = (float*)d_out;

  char* ws = (char*)d_ws;
  size_t off = 0;
  __bf16* Wqkv  = (__bf16*)(ws + off); off += (size_t)256 * 768 * 2;
  __bf16* Wproj = (__bf16*)(ws + off); off += (size_t)256 * 256 * 2;
  __bf16* Wfc1  = (__bf16*)(ws + off); off += (size_t)256 * 1024 * 2;
  __bf16* Wfc2  = (__bf16*)(ws + off); off += (size_t)1024 * 256 * 2;
  __bf16* hw    = (__bf16*)(ws + off); off += (size_t)MROWS * CC * 2;      // reused as attn_out
  __bf16* qkvb  = (__bf16*)(ws + off); off += (size_t)MROWS * 768 * 2;     // start reused as h2
  __bf16* attn_out = hw;
  __bf16* h2 = qkvb;

  pack_w_kernel<<<(256 * 768 + 255) / 256, 256, 0, stream>>>(qkv_w, Wqkv, 256, 768);
  pack_w_kernel<<<(256 * 256 + 255) / 256, 256, 0, stream>>>(proj_w, Wproj, 256, 256);
  pack_w_kernel<<<(256 * 1024 + 255) / 256, 256, 0, stream>>>(fc1_w, Wfc1, 256, 1024);
  pack_w_kernel<<<(1024 * 256 + 255) / 256, 256, 0, stream>>>(fc2_w, Wfc2, 1024, 256);

  ln_kernel<<<MROWS / 8, 256, 0, stream>>>(x, n1w, n1b, hw, SHIFT_);

  qkv_gemm_kernel<<<MROWS / 32, 256, 0, stream>>>(hw, Wqkv, qkv_b, qkvb);

  attn_kernel<<<BB * NWIN * NHEADS, 128, 0, stream>>>(qkvb, bias_tb, rel, attn_out);

  proj_gemm_kernel<<<MROWS / 32, 256, 0, stream>>>(attn_out, Wproj, proj_b, x, out);

  ln_kernel<<<MROWS / 8, 256, 0, stream>>>(out, n2w, n2b, h2, 0);

  mlp_kernel<<<MROWS / 16, 256, 0, stream>>>(h2, Wfc1, fc1_b, Wfc2, fc2_b, out);
}
